// VectorQuantizer_31980326486406
// MI455X (gfx1250) — matrix operand order repaired and checked by reading, not compiled, not run on hardware
//
#include <hip/hip_runtime.h>

typedef __attribute__((ext_vector_type(16))) __bf16 v16bf;
typedef __attribute__((ext_vector_type(8)))  __bf16 v8bf;
typedef __attribute__((ext_vector_type(8)))  float  v8f;

#define N_E     16384
#define E_DIM   256
#define NVEC    32768        // 8 * 64 * 64
#define HW      4096         // 64 * 64
#define NTILES  (N_E / 16)   // 1024 codebook tiles of 16 codes
#define MBLOCKS (NVEC / 32)  // 1024 z blocks of 32 vectors (2 WMMA M-tiles)
#define KCHUNKS (E_DIM / 32) // 8 k-chunks of 32 per WMMA

// -------- 1) codebook squared norms --------
__global__ void vq_enorm(const float* __restrict__ cb, float* __restrict__ enorm) {
    int code = blockIdx.x * blockDim.x + threadIdx.x;
    if (code >= N_E) return;
    const float4* p = (const float4*)(cb + code * E_DIM);
    float s = 0.f;
#pragma unroll
    for (int i = 0; i < E_DIM / 4; ++i) {
        float4 v = p[i];
        s += v.x * v.x + v.y * v.y + v.z * v.z + v.w * v.w;
    }
    enorm[code] = s;
}

// -------- 2) codebook -> bf16 B-fragments in WMMA lane layout --------
// B is 32x16 (KxN) per chunk. Lane L: n = L%16, half = L/16, element i <-> K = half*16 + i.
// Fragment (t, c) stored as 32 lanes x 16 bf16 contiguous -> hot loop loads are coalesced b128s.
__global__ void vq_bfrag(const float* __restrict__ cb, v16bf* __restrict__ bfrag) {
    int tid = blockIdx.x * blockDim.x + threadIdx.x;   // 0 .. N_E*E_DIM/16 - 1
    if (tid >= N_E * E_DIM / 16) return;
    int lane = tid & 31;
    int c    = (tid >> 5) & 7;
    int t    = tid >> 8;
    int n    = lane & 15;
    int half = lane >> 4;
    const float* src = cb + (t * 16 + n) * E_DIM + c * 32 + half * 16;
    v16bf v;
#pragma unroll
    for (int i = 0; i < 16; ++i) v[i] = (__bf16)src[i];
    bfrag[tid] = v;
}

// -------- 3) z [B,C,H,W] fp32 -> row-major bf16 A matrix [n][k], n = b*HW + hw, k = c --------
__global__ void vq_abf(const float* __restrict__ z, __bf16* __restrict__ abf) {
    int tid = blockIdx.x * blockDim.x + threadIdx.x;   // 0 .. NVEC*E_DIM - 1
    int k  = tid & 255;
    int n  = tid >> 8;
    int b  = n >> 12;
    int hw = n & 4095;
    abf[tid] = (__bf16)z[(b * E_DIM + k) * HW + hw];
}

// -------- 4) main: fused distance GEMM + argmin, M-blocked x2, codebook split x2 --------
// One wave per (32-row M-block, codebook half). A for both M-tiles held in registers;
// each B fragment feeds TWO WMMAs (32 FLOP / L0 byte). B is explicitly double-buffered
// (bcur/bnext) so the next fragment's loads are in flight while the current pair of
// WMMAs executes -- including across the tile boundary (first chunk of tile t+1).
// A k order used here (contiguous, same as B): lane L: m = L%16, half = L/16; element i: K=half*16+i.
// C/D layout: lane L: col n = L%16; VGPR r: row m = (L<16 ? r : 8+r).
__global__ void __launch_bounds__(256)
vq_argmin(const __bf16* __restrict__ abf, const v16bf* __restrict__ bfrag,
          const float* __restrict__ enorm, float* __restrict__ pbest,
          int* __restrict__ pidx) {
    int lane  = threadIdx.x & 31;
    int wv    = (blockIdx.x * blockDim.x + threadIdx.x) >> 5;  // 0 .. 2047
    int mblk  = wv >> 1;                                       // 0 .. MBLOCKS-1
    int nhalf = wv & 1;                                        // codebook half
    int r16   = lane & 15;
    int half  = lane >> 4;

    v16bf afrag[2][KCHUNKS];
#pragma unroll
    for (int mt = 0; mt < 2; ++mt) {
        const __bf16* ap = abf + (mblk * 32 + mt * 16 + r16) * E_DIM;
#pragma unroll
        for (int c = 0; c < KCHUNKS; ++c) {
            v8bf lo = *(const v8bf*)(ap + c * 32 + half * 16);      // K = c*32 + half*16 + 0..7
            v8bf hi = *(const v8bf*)(ap + c * 32 + half * 16 + 8);  // K = c*32 + half*16 + 8..15
            afrag[mt][c] = __builtin_shufflevector(lo, hi, 0, 1, 2, 3, 4, 5, 6, 7,
                                                           8, 9, 10, 11, 12, 13, 14, 15);
        }
    }

    float best[2][8];
    int   bidx[2][8];
#pragma unroll
    for (int r = 0; r < 8; ++r) {
        best[0][r] = 3.4e38f; best[1][r] = 3.4e38f;
        bidx[0][r] = 0;       bidx[1][r] = 0;
    }

    int tbeg = nhalf * (NTILES / 2);
    int tend = tbeg + (NTILES / 2);

    const v16bf* bp = bfrag + (size_t)tbeg * KCHUNKS * 32 + lane;
    v16bf bcur = bp[0];                      // prologue: first fragment in flight

    for (int t = tbeg; t < tend; ++t) {
        float en = enorm[t * 16 + r16];      // in flight during the WMMAs below

        v8f acc0 = {0.f, 0.f, 0.f, 0.f, 0.f, 0.f, 0.f, 0.f};
        v8f acc1 = {0.f, 0.f, 0.f, 0.f, 0.f, 0.f, 0.f, 0.f};
#pragma unroll
        for (int c = 0; c < KCHUNKS; ++c) {
            v16bf bnext = bcur;
            if (c + 1 < KCHUNKS)      bnext = bp[(c + 1) * 32];      // next chunk, this tile
            else if (t + 1 < tend)    bnext = bp[KCHUNKS * 32];      // chunk 0, next tile
            acc0 = __builtin_amdgcn_wmma_f32_16x16x32_bf16(
                false, afrag[0][c], false, bcur, (short)0, acc0, false, false);
            acc1 = __builtin_amdgcn_wmma_f32_16x16x32_bf16(
                false, afrag[1][c], false, bcur, (short)0, acc1, false, false);
            bcur = bnext;
        }
        bp += KCHUNKS * 32;

        int code = t * 16 + r16;             // column index handled by this lane
#pragma unroll
        for (int r = 0; r < 8; ++r) {
            float d0 = en - 2.0f * acc0[r];  // ||x||^2 dropped: row-constant
            float d1 = en - 2.0f * acc1[r];
            if (d0 < best[0][r]) { best[0][r] = d0; bidx[0][r] = code; }
            if (d1 < best[1][r]) { best[1][r] = d1; bidx[1][r] = code; }
        }
    }

    // reduce over the 16 columns held by each 16-lane half; tie-break to lowest index
#pragma unroll
    for (int mt = 0; mt < 2; ++mt) {
#pragma unroll
        for (int r = 0; r < 8; ++r) {
#pragma unroll
            for (int off = 1; off < 16; off <<= 1) {
                float ov = __shfl_xor(best[mt][r], off, 32);
                int   oi = __shfl_xor(bidx[mt][r], off, 32);
                if (ov < best[mt][r] || (ov == best[mt][r] && oi < bidx[mt][r])) {
                    best[mt][r] = ov; bidx[mt][r] = oi;
                }
            }
            if (r16 == 0) {
                int row = mblk * 32 + mt * 16 + half * 8 + r;  // half 0 -> rows 0..7, half 1 -> 8..15
                pbest[nhalf * NVEC + row] = best[mt][r];
                pidx [nhalf * NVEC + row] = bidx[mt][r];
            }
        }
    }
}

// -------- 5) merge the two codebook-half candidates per row --------
__global__ void vq_merge(const float* __restrict__ pbest, const int* __restrict__ pidx,
                         int* __restrict__ idxout) {
    int n = blockIdx.x * blockDim.x + threadIdx.x;   // 0 .. NVEC-1
    float d0 = pbest[n], d1 = pbest[NVEC + n];
    int   i0 = pidx[n],  i1 = pidx[NVEC + n];
    idxout[n] = (d1 < d0) ? i1 : i0;                 // tie -> i0 (lower index: half 0)
}

// -------- 6) gather codebook rows back to [B, C, H, W] --------
__global__ void vq_gather(const float* __restrict__ cb, const int* __restrict__ idx,
                          float* __restrict__ out) {
    int tid = blockIdx.x * blockDim.x + threadIdx.x;   // linear index into [B,C,H,W]
    int hw = tid & 4095;
    int c  = (tid >> 12) & 255;
    int b  = tid >> 20;
    int n  = b * HW + hw;
    out[tid] = cb[idx[n] * E_DIM + c];
}

extern "C" void kernel_launch(void* const* d_in, const int* in_sizes, int n_in,
                              void* d_out, int out_size, void* d_ws, size_t ws_size,
                              hipStream_t stream) {
    const float* z  = (const float*)d_in[0];
    const float* cb = (const float*)d_in[1];
    float* out = (float*)d_out;

    char*   ws    = (char*)d_ws;
    __bf16* abf   = (__bf16*)ws;                          // 16 MiB
    v16bf*  bfrag = (v16bf*)(ws + 16777216);              //  8 MiB
    float*  enorm = (float*)(ws + 25165824);              // 64 KiB
    int*    idx   = (int*)  (ws + 25231360);              // 128 KiB
    float*  pbest = (float*)(ws + 25362432);              // 256 KiB (2 x NVEC)
    int*    pidx  = (int*)  (ws + 25624576);              // 256 KiB (2 x NVEC)

    vq_enorm <<<N_E / 256, 256, 0, stream>>>(cb, enorm);
    vq_bfrag <<<(N_E * E_DIM / 16) / 256, 256, 0, stream>>>(cb, bfrag);
    vq_abf   <<<(NVEC * E_DIM) / 256, 256, 0, stream>>>(z, abf);
    vq_argmin<<<(MBLOCKS * 2) / 8, 256, 0, stream>>>(abf, bfrag, enorm, pbest, pidx);
    vq_merge <<<NVEC / 256, 256, 0, stream>>>(pbest, pidx, idx);
    vq_gather<<<(NVEC * E_DIM) / 256, 256, 0, stream>>>(cb, idx, out);
}
